// NystromAttention_8040178778175
// MI455X (gfx1250) — matrix operand order repaired and checked by reading, not compiled, not run on hardware
//
#include <hip/hip_runtime.h>
#include <stdint.h>

#define DEVFN __device__ __forceinline__

typedef __bf16 bf16;
typedef __attribute__((ext_vector_type(16))) __bf16 v16bf;
typedef __attribute__((ext_vector_type(8)))  __bf16 v8bf;
typedef __attribute__((ext_vector_type(8)))  float  v8f;
typedef __attribute__((ext_vector_type(4)))  float  v4f;
typedef __attribute__((ext_vector_type(4)))  unsigned int v4u;
typedef __attribute__((ext_vector_type(4)))  int    v4i;
typedef __attribute__((ext_vector_type(8)))  int    v8i;

constexpr int NB  = 32;    // batch*heads
constexpr int SEQ = 8192;  // sequence length
constexpr int HD  = 64;    // head dim
constexpr int NL  = 64;    // landmarks
constexpr float SCALE = 0.125f;  // 1/sqrt(64)

// ---------------- WMMA + fragment helpers ----------------

DEVFN v8f wmma_bf16(v16bf a, v16bf b, v8f c) {
  // v_wmma_f32_16x16x32_bf16  (neg_a, A, neg_b, B, c_mod, C, reuse_a, reuse_b)
  return __builtin_amdgcn_wmma_f32_16x16x32_bf16(false, a, false, b, (short)0, c, false, false);
}

DEVFN v16bf join8(v8bf lo, v8bf hi) {
  return __builtin_shufflevector(lo, hi, 0,1,2,3,4,5,6,7,8,9,10,11,12,13,14,15);
}

// B-fragment: 16 contiguous fp32 -> bf16 (lane holds K-run of 16)
DEVFN v16bf cvt_bfrag(const float* p) {
  const v4f* p4 = (const v4f*)p;
  v4f x0 = p4[0], x1 = p4[1], x2 = p4[2], x3 = p4[3];
  v16bf r;
#pragma unroll
  for (int i = 0; i < 4; ++i) {
    r[i]      = (bf16)x0[i];
    r[4 + i]  = (bf16)x1[i];
    r[8 + i]  = (bf16)x2[i];
    r[12 + i] = (bf16)x3[i];
  }
  return r;
}

// A-fragment from fp32: two 8-float runs (K = 16h..16h+7 and 16h+8..16h+15), scaled
DEVFN v16bf cvt_afrag(const float* lo, const float* hi, float sc) {
  const v4f* l4 = (const v4f*)lo;
  const v4f* h4 = (const v4f*)hi;
  v4f a0 = l4[0], a1 = l4[1], b0 = h4[0], b1 = h4[1];
  v16bf r;
#pragma unroll
  for (int i = 0; i < 4; ++i) {
    r[i]      = (bf16)(a0[i] * sc);
    r[4 + i]  = (bf16)(a1[i] * sc);
    r[8 + i]  = (bf16)(b0[i] * sc);
    r[12 + i] = (bf16)(b1[i] * sc);
  }
  return r;
}

DEVFN float redmax16(float v) {   // reduce across 16-lane half (wave32)
  v = fmaxf(v, __shfl_xor(v, 1, 32));
  v = fmaxf(v, __shfl_xor(v, 2, 32));
  v = fmaxf(v, __shfl_xor(v, 4, 32));
  v = fmaxf(v, __shfl_xor(v, 8, 32));
  return v;
}
DEVFN float redsum16(float v) {
  v += __shfl_xor(v, 1, 32);
  v += __shfl_xor(v, 2, 32);
  v += __shfl_xor(v, 4, 32);
  v += __shfl_xor(v, 8, 32);
  return v;
}

// ---------------- TDM: async 64x64 bf16 tile -> LDS (padded rows) ----------------
// D# per CDNA5 ISA §8: group0 = {flags, lds_addr, global_addr lo/hi|type=2},
// group1 = {mask/dsize/pad, dims, tile dims, strides}. Rows are 128B in memory;
// pad_enable inserts 16B per 128B -> LDS row stride 144B (36 banks, conflict-free
// for 16-row B-fragment reads, 16B-aligned for ds_load_b128).

#define TROW 72   // padded LDS row stride in bf16 elements (144 B)

DEVFN void tdm_load_bf16_64x64(const bf16* gsrc, unsigned lds_byte_off) {
  unsigned long long ga = (unsigned long long)(uintptr_t)gsrc;
  v4u g0;
  g0[0] = 1u;                                        // count=1 (valid user D#)
  g0[1] = lds_byte_off;                              // lds_addr
  g0[2] = (unsigned)(ga & 0xFFFFFFFFull);            // global_addr[31:0]
  g0[3] = (unsigned)((ga >> 32) & 0x01FFFFFFull)     // global_addr[56:32]
        | 0x80000000u;                               // type=2 ("image")
  v8i g1;
  g1[0] = (int)((1u << 16)        // data_size = 1 -> 2 bytes
              | (1u << 20)        // pad_enable
              | (4u << 22)        // pad_interval: 32 DWORDs (one 128B row)
              | (3u << 25));      // pad_amount: 4 DWORDs (16B)
  g1[1] = (int)(64u << 16);       // tensor_dim0 = 64 (low 16 -> [31:16])
  g1[2] = (int)(64u << 16);       // tensor_dim0 hi=0 | tensor_dim1 low16 = 64
  g1[3] = (int)(64u << 16);       // tensor_dim1 hi=0 | tile_dim0 = 64
  g1[4] = 64;                     // tile_dim1 = 64 | tile_dim2 = 0
  g1[5] = 64;                     // tensor_dim0_stride = 64 (low 32)
  g1[6] = 0;                      // stride0 hi | stride1 low16
  g1[7] = 0;
  v4i z = {0, 0, 0, 0};
#if __clang_major__ >= 23
  v8i z8 = {0, 0, 0, 0, 0, 0, 0, 0};
  __builtin_amdgcn_tensor_load_to_lds(g0, g1, z, z, z8, 0);
#else
  __builtin_amdgcn_tensor_load_to_lds(g0, g1, z, z, 0);
#endif
}

// ---------------- 1) landmark pooling -> bf16 (q side pre-scaled) ----------------

__global__ void nys_pool(const float* __restrict__ q, const float* __restrict__ k,
                         bf16* __restrict__ qls, bf16* __restrict__ kl) {
  int blk = blockIdx.x;
  int b = blk >> 6, n = blk & 63;
  int d = threadIdx.x;  // 64 threads
  const float* qp = q + (((size_t)b * SEQ) + (size_t)n * 128) * HD + d;
  const float* kp = k + (((size_t)b * SEQ) + (size_t)n * 128) * HD + d;
  float sq = 0.f, sk = 0.f;
#pragma unroll 4
  for (int j = 0; j < 128; ++j) {
    sq += qp[(size_t)j * HD];
    sk += kp[(size_t)j * HD];
  }
  qls[(size_t)(b * NL + n) * HD + d] = (bf16)(sq * (SCALE / 128.f));
  kl [(size_t)(b * NL + n) * HD + d] = (bf16)(sk * (1.f / 128.f));
}

// ---------------- 2) V -> bf16 transposed [b][d][s] ----------------

__global__ __launch_bounds__(256) void nys_vtrans(const float* __restrict__ v, bf16* __restrict__ vT) {
  int b = blockIdx.y;
  int s0 = blockIdx.x * 128;
  __shared__ bf16 tile[64 * 132];
  int t = threadIdx.x;
  const float* src = v + ((size_t)b * SEQ + s0) * HD;
#pragma unroll
  for (int i = 0; i < 32; ++i) {
    int idx = i * 256 + t;        // [128 s][64 d], s-major
    int s = idx >> 6, d = idx & 63;
    tile[d * 132 + s] = (bf16)src[idx];
  }
  __syncthreads();
  bf16* dst = vT + (size_t)b * HD * SEQ + s0;
#pragma unroll
  for (int i = 0; i < 32; ++i) {
    int idx = i * 256 + t;        // [64 d][128 s], d-major
    int d = idx >> 7, s = idx & 127;
    dst[(size_t)d * SEQ + s] = tile[d * 132 + s];
  }
}

// ---------------- 3) kernel_2 softmax + Newton-Schulz pinv (fp32, LDS) ----------------

#define PM 68  // padded row stride (272B: 16B aligned, conflict-free v4f)

DEVFN void mm64(const float* A, const float* Bm, float* C, int r4, int c4) {
  float acc[4][4] = {};
#pragma unroll 4
  for (int kk = 0; kk < 64; ++kk) {
    float a0 = A[(r4 + 0) * PM + kk];
    float a1 = A[(r4 + 1) * PM + kk];
    float a2 = A[(r4 + 2) * PM + kk];
    float a3 = A[(r4 + 3) * PM + kk];
    v4f bv = *(const v4f*)(Bm + kk * PM + c4);
#pragma unroll
    for (int j = 0; j < 4; ++j) {
      acc[0][j] += a0 * bv[j];
      acc[1][j] += a1 * bv[j];
      acc[2][j] += a2 * bv[j];
      acc[3][j] += a3 * bv[j];
    }
  }
#pragma unroll
  for (int i = 0; i < 4; ++i) {
    v4f r;
#pragma unroll
    for (int j = 0; j < 4; ++j) r[j] = acc[i][j];
    *(v4f*)(C + (r4 + i) * PM + c4) = r;
  }
}

__global__ __launch_bounds__(256) void nys_pinv(const bf16* __restrict__ qls,
                                                const bf16* __restrict__ kl,
                                                bf16* __restrict__ k2invT) {
  int b = blockIdx.x, t = threadIdx.x;
  __shared__ float smem[5 * 64 * PM];
  __shared__ float cbuf[64];
  __shared__ float sinv;
  float* Km = smem;
  float* Vm = smem + 64 * PM;
  float* KV = smem + 2 * 64 * PM;
  float* TA = smem + 3 * 64 * PM;
  float* TB = smem + 4 * 64 * PM;
  bf16* qb = (bf16*)TA;   // staging aliases (dead before TA/TB used)
  bf16* kb = (bf16*)TB;

  for (int i = t; i < 4096; i += 256) {
    qb[i] = qls[(size_t)b * 4096 + i];
    kb[i] = kl [(size_t)b * 4096 + i];
  }
  __syncthreads();

  int r4 = (t >> 4) * 4, c4 = (t & 15) * 4;
  {  // G[n][m] = sum_d qls[n][d]*kl[m][d]  (qls already scaled)
    float acc[4][4] = {};
    for (int d = 0; d < 64; ++d) {
      float a0 = (float)qb[(r4 + 0) * 64 + d], a1 = (float)qb[(r4 + 1) * 64 + d];
      float a2 = (float)qb[(r4 + 2) * 64 + d], a3 = (float)qb[(r4 + 3) * 64 + d];
      float b0 = (float)kb[(c4 + 0) * 64 + d], b1 = (float)kb[(c4 + 1) * 64 + d];
      float b2 = (float)kb[(c4 + 2) * 64 + d], b3 = (float)kb[(c4 + 3) * 64 + d];
      acc[0][0] += a0 * b0; acc[0][1] += a0 * b1; acc[0][2] += a0 * b2; acc[0][3] += a0 * b3;
      acc[1][0] += a1 * b0; acc[1][1] += a1 * b1; acc[1][2] += a1 * b2; acc[1][3] += a1 * b3;
      acc[2][0] += a2 * b0; acc[2][1] += a2 * b1; acc[2][2] += a2 * b2; acc[2][3] += a2 * b3;
      acc[3][0] += a3 * b0; acc[3][1] += a3 * b1; acc[3][2] += a3 * b2; acc[3][3] += a3 * b3;
    }
    __syncthreads();  // staging consumed
#pragma unroll
    for (int i = 0; i < 4; ++i)
#pragma unroll
      for (int j = 0; j < 4; ++j) Km[(r4 + i) * PM + c4 + j] = acc[i][j];
  }
  __syncthreads();

  if (t < 64) {  // row softmax
    float mx = -1e30f;
    for (int m = 0; m < 64; ++m) mx = fmaxf(mx, Km[t * PM + m]);
    float s = 0.f;
    for (int m = 0; m < 64; ++m) { float e = __expf(Km[t * PM + m] - mx); Km[t * PM + m] = e; s += e; }
    float inv = 1.f / s;
    for (int m = 0; m < 64; ++m) Km[t * PM + m] *= inv;
  }
  __syncthreads();
  if (t < 64) {  // column sums
    float cs = 0.f;
    for (int n = 0; n < 64; ++n) cs += Km[n * PM + t];
    cbuf[t] = cs;
  }
  __syncthreads();
  if (t == 0) {
    float mx = 0.f;
    for (int m = 0; m < 64; ++m) mx = fmaxf(mx, cbuf[m]);
    sinv = 1.f / mx;
  }
  __syncthreads();
  for (int i = t; i < 4096; i += 256) {
    int n = i >> 6, m = i & 63;
    Vm[n * PM + m] = Km[m * PM + n] * sinv;   // V0 = K^T / max_col_sum
  }
  __syncthreads();

  for (int iter = 0; iter < 6; ++iter) {
    mm64(Km, Vm, KV, r4, c4); __syncthreads();
    for (int i = t; i < 4096; i += 256) { int n = i >> 6, m = i & 63; TA[n * PM + m] = ((n == m) ? 7.f : 0.f) - KV[n * PM + m]; }
    __syncthreads();
    mm64(KV, TA, TB, r4, c4); __syncthreads();
    for (int i = t; i < 4096; i += 256) { int n = i >> 6, m = i & 63; TB[n * PM + m] = ((n == m) ? 15.f : 0.f) - TB[n * PM + m]; }
    __syncthreads();
    mm64(KV, TB, TA, r4, c4); __syncthreads();
    for (int i = t; i < 4096; i += 256) { int n = i >> 6, m = i & 63; TA[n * PM + m] = ((n == m) ? 13.f : 0.f) - TA[n * PM + m]; }
    __syncthreads();
    mm64(Vm, TA, TB, r4, c4); __syncthreads();
    for (int i = t; i < 4096; i += 256) Vm[i / 64 * PM + i % 64] = 0.25f * TB[i / 64 * PM + i % 64];
    __syncthreads();
  }
  // store transposed: k2invT[n][m] = V[m][n]  (B-fragment friendly)
  for (int i = t; i < 4096; i += 256) {
    int n = i >> 6, m = i & 63;
    k2invT[(size_t)b * 4096 + n * 64 + m] = (bf16)Vm[m * PM + n];
  }
}

// ---------------- 4a) logits L[b][m][s] = qls . k  (WMMA) ----------------

__global__ __launch_bounds__(256) void nys_k3a(const bf16* __restrict__ qls,
                                               const float* __restrict__ kmat,
                                               bf16* __restrict__ Lb) {
  int b = blockIdx.y;
  int t = threadIdx.x, w = t >> 5, ln = t & 31, lc = ln & 15, h = ln >> 4;
  int m0 = (w & 3) * 16;
  int sbase = blockIdx.x * 512 + (w >> 2) * 256;

  const bf16* qp = qls + ((size_t)(b * NL) + m0 + lc) * HD;
  v16bf a0 = join8(*(const v8bf*)(qp + 16 * h),      *(const v8bf*)(qp + 8 + 16 * h));
  v16bf a1 = join8(*(const v8bf*)(qp + 32 + 16 * h), *(const v8bf*)(qp + 40 + 16 * h));

  for (int st = 0; st < 16; ++st) {
    int s = sbase + st * 16;
    const float* kp = kmat + ((size_t)b * SEQ + s + lc) * HD + 16 * h;
    __builtin_prefetch(kp + 16 * HD, 0, 0);   // global_prefetch_b8 for next tile
    v16bf b0 = cvt_bfrag(kp);
    v16bf b1 = cvt_bfrag(kp + 32);
    v8f c = {};
    c = wmma_bf16(a0, b0, c);
    c = wmma_bf16(a1, b1, c);
    bf16* lp = Lb + ((size_t)(b * NL + m0 + 8 * h)) * SEQ + s + lc;
#pragma unroll
    for (int r = 0; r < 8; ++r) lp[(size_t)r * SEQ] = (bf16)c[r];
  }
}

// ---------------- 4b) per-row max of L ----------------

__global__ __launch_bounds__(256) void nys_lmax(const bf16* __restrict__ Lb, float* __restrict__ Lmax) {
  int row = blockIdx.x, t = threadIdx.x;
  const bf16* p = Lb + (size_t)row * SEQ;
  float m = -1e30f;
  for (int i = t; i < SEQ; i += 256) m = fmaxf(m, (float)p[i]);
  __shared__ float red[256];
  red[t] = m;
  __syncthreads();
  for (int off = 128; off; off >>= 1) {
    if (t < off) red[t] = fmaxf(red[t], red[t + off]);
    __syncthreads();
  }
  if (t == 0) Lmax[row] = red[0];
}

// ---------------- 4c) P = exp(L - max);  partial O = P @ V  (WMMA) ----------------

__global__ __launch_bounds__(256) void nys_k3b(const bf16* __restrict__ Lb,
                                               const float* __restrict__ Lmax,
                                               const bf16* __restrict__ vT,
                                               float* __restrict__ pO,
                                               float* __restrict__ psum) {
  int b = blockIdx.y, ch = blockIdx.x;  // 8 chunks of 1024
  int t = threadIdx.x, w = t >> 5, ln = t & 31, lc = ln & 15, h = ln >> 4;
  int m0 = (w & 3) * 16, half = w >> 2;
  int sbase = ch * 1024 + half * 512;
  int mrow = m0 + lc;
  float mmax = Lmax[b * NL + mrow];
  const bf16* lrow = Lb + ((size_t)(b * NL) + mrow) * SEQ;
  const bf16* vbase = vT + (size_t)b * HD * SEQ;

  v8f c0 = {}, c1 = {}, c2 = {}, c3 = {};
  float sume = 0.f;
  for (int it = 0; it < 16; ++it) {
    int s = sbase + it * 32;
    v8bf lo = *(const v8bf*)(lrow + s + 16 * h);
    v8bf hi = *(const v8bf*)(lrow + s + 8 + 16 * h);
    v16bf a;
#pragma unroll
    for (int j = 0; j < 8; ++j) {
      float e0 = __expf((float)lo[j] - mmax); sume += e0; a[j]     = (bf16)e0;
      float e1 = __expf((float)hi[j] - mmax); sume += e1; a[j + 8] = (bf16)e1;
    }
    const bf16* vp = vbase + (size_t)lc * SEQ + s + 16 * h;
    v16bf b0 = *(const v16bf*)(vp);
    v16bf b1 = *(const v16bf*)(vp + (size_t)16 * SEQ);
    v16bf b2 = *(const v16bf*)(vp + (size_t)32 * SEQ);
    v16bf b3 = *(const v16bf*)(vp + (size_t)48 * SEQ);
    c0 = wmma_bf16(a, b0, c0);
    c1 = wmma_bf16(a, b1, c1);
    c2 = wmma_bf16(a, b2, c2);
    c3 = wmma_bf16(a, b3, c3);
  }
  sume += __shfl_xor(sume, 16, 32);
  int slot = b * 16 + ch * 2 + half;
  if (ln < 16) psum[slot * 64 + mrow] = sume;
  float* po = pO + ((size_t)slot * 64 + m0 + 8 * h) * 64 + lc;
#pragma unroll
  for (int r = 0; r < 8; ++r) {
    po[(size_t)r * 64 + 0]  = c0[r];
    po[(size_t)r * 64 + 16] = c1[r];
    po[(size_t)r * 64 + 32] = c2[r];
    po[(size_t)r * 64 + 48] = c3[r];
  }
}

// ---------------- 4d) combine partials -> kernel_3 transposed bf16 ----------------

__global__ __launch_bounds__(256) void nys_k3c(const float* __restrict__ pO,
                                               const float* __restrict__ psum,
                                               bf16* __restrict__ k3T) {
  int b = blockIdx.x, t = threadIdx.x;
  for (int o = t; o < 4096; o += 256) {
    int m = o >> 6, d = o & 63;
    float acc = 0.f, ss = 0.f;
#pragma unroll
    for (int sl = 0; sl < 16; ++sl) {
      int gs = b * 16 + sl;
      acc += pO[((size_t)gs * 64 + m) * 64 + d];
      ss  += psum[gs * 64 + m];
    }
    k3T[(size_t)b * 4096 + d * 64 + m] = (bf16)(acc / ss);
  }
}

// ---------------- 5) fused: softmax(q kl^T) @ k2inv @ kernel_3 -> out ----------------
// The three per-batch 64x64 bf16 operand matrices are TDM-loaded into LDS once per
// block (async, TENSORcnt-tracked) instead of 8 waves x 3 x 8KB of global loads.

__global__ __launch_bounds__(256) void nys_fuse(const float* __restrict__ q,
                                                const bf16* __restrict__ kl,
                                                const bf16* __restrict__ k2invT,
                                                const bf16* __restrict__ k3T,
                                                float* __restrict__ out) {
  int b = blockIdx.y;
  int t = threadIdx.x, w = t >> 5, ln = t & 31, lc = ln & 15, h = ln >> 4;
  int s0 = blockIdx.x * 128 + w * 16;
  __shared__ bf16 Bmat[3 * 64 * TROW];     // TDM-padded rows: 144B stride
  __shared__ bf16 Pbuf[8 * 16 * 88];       // wave-private staging, conflict-free b128
  bf16* P = Pbuf + w * 16 * 88;

  if (t < 32) {  // wave 0 issues the three async tensor loads
    unsigned base = (unsigned)(uintptr_t)(&Bmat[0]);
    tdm_load_bf16_64x64(kl     + (size_t)b * 4096, base);
    tdm_load_bf16_64x64(k2invT + (size_t)b * 4096, base + 64 * TROW * 2);
    tdm_load_bf16_64x64(k3T    + (size_t)b * 4096, base + 2 * 64 * TROW * 2);
    __builtin_amdgcn_s_wait_tensorcnt(0);
  }

  int srow = s0 + lc;
  const float* qp = q + ((size_t)b * SEQ + srow) * HD;
  v16bf a0 = cvt_afrag(qp + 16 * h,      qp + 8 + 16 * h, SCALE);
  v16bf a1 = cvt_afrag(qp + 32 + 16 * h, qp + 40 + 16 * h, SCALE);

  __syncthreads();  // LDS B-matrices ready for all waves

  const bf16* klb = &Bmat[0];
  const bf16* k2b = &Bmat[64 * TROW];
  const bf16* k3p = &Bmat[2 * 64 * TROW];

  v8f c[4];
#pragma unroll
  for (int n = 0; n < 4; ++n) {
    const bf16* bp = klb + (n * 16 + lc) * TROW + 16 * h;
    v8f acc = {};
    acc = wmma_bf16(a0, *(const v16bf*)bp, acc);
    acc = wmma_bf16(a1, *(const v16bf*)(bp + 32), acc);
    c[n] = acc;
  }

  // softmax over the 64 landmark columns
#pragma unroll
  for (int r = 0; r < 8; ++r) {
    float m = fmaxf(fmaxf(c[0][r], c[1][r]), fmaxf(c[2][r], c[3][r]));
    m = redmax16(m);
    float s = 0.f;
#pragma unroll
    for (int n = 0; n < 4; ++n) { float e = __expf(c[n][r] - m); c[n][r] = e; s += e; }
    s = redsum16(s);
    float inv = 1.f / s;
#pragma unroll
    for (int n = 0; n < 4; ++n) c[n][r] *= inv;
  }

  // stage P to LDS (C-layout -> A-layout)
#pragma unroll
  for (int r = 0; r < 8; ++r)
#pragma unroll
    for (int n = 0; n < 4; ++n)
      P[(r + 8 * h) * 88 + n * 16 + lc] = (bf16)c[n][r];

  v16bf pa0 = join8(*(const v8bf*)(P + lc * 88 + 16 * h),      *(const v8bf*)(P + lc * 88 + 8 + 16 * h));
  v16bf pa1 = join8(*(const v8bf*)(P + lc * 88 + 32 + 16 * h), *(const v8bf*)(P + lc * 88 + 40 + 16 * h));

#pragma unroll
  for (int n = 0; n < 4; ++n) {
    const bf16* bp = k2b + (n * 16 + lc) * TROW + 16 * h;
    v8f acc = {};
    acc = wmma_bf16(pa0, *(const v16bf*)bp, acc);
    acc = wmma_bf16(pa1, *(const v16bf*)(bp + 32), acc);
    c[n] = acc;
  }

  // stage x1 to LDS (DS ops are in-order per wave: prior loads precede these stores)
#pragma unroll
  for (int r = 0; r < 8; ++r)
#pragma unroll
    for (int n = 0; n < 4; ++n)
      P[(r + 8 * h) * 88 + n * 16 + lc] = (bf16)c[n][r];

  v16bf xa0 = join8(*(const v8bf*)(P + lc * 88 + 16 * h),      *(const v8bf*)(P + lc * 88 + 8 + 16 * h));
  v16bf xa1 = join8(*(const v8bf*)(P + lc * 88 + 32 + 16 * h), *(const v8bf*)(P + lc * 88 + 40 + 16 * h));

#pragma unroll
  for (int n = 0; n < 4; ++n) {
    const bf16* bp = k3p + (n * 16 + lc) * TROW + 16 * h;
    v8f acc = {};
    acc = wmma_bf16(xa0, *(const v16bf*)bp, acc);
    acc = wmma_bf16(xa1, *(const v16bf*)(bp + 32), acc);
    c[n] = acc;
  }

  float* op = out + ((size_t)b * SEQ + s0 + 8 * h) * HD + lc;
#pragma unroll
  for (int r = 0; r < 8; ++r)
#pragma unroll
    for (int n = 0; n < 4; ++n)
      op[(size_t)r * HD + n * 16] = c[n][r];
}

// ---------------- host-side launch ----------------

extern "C" void kernel_launch(void* const* d_in, const int* in_sizes, int n_in,
                              void* d_out, int out_size, void* d_ws, size_t ws_size,
                              hipStream_t stream) {
  (void)in_sizes; (void)n_in; (void)out_size; (void)ws_size;
  const float* q = (const float*)d_in[0];
  const float* k = (const float*)d_in[1];
  const float* v = (const float*)d_in[2];
  float* out = (float*)d_out;

  char* w = (char*)d_ws;
  constexpr size_t SMALL = (size_t)NB * NL * HD * sizeof(bf16);          // 256 KB
  constexpr size_t OFF_QLS  = 0;
  constexpr size_t OFF_KL   = OFF_QLS + SMALL;
  constexpr size_t OFF_K2   = OFF_KL + SMALL;
  constexpr size_t OFF_K3   = OFF_K2 + SMALL;
  constexpr size_t OFF_VT   = (size_t)1 << 20;                           // 1 MB, aligned
  constexpr size_t SZ_VT    = (size_t)NB * HD * SEQ * sizeof(bf16);      // 32 MB
  constexpr size_t OFF_L    = OFF_VT + SZ_VT;
  constexpr size_t SZ_L     = (size_t)NB * NL * SEQ * sizeof(bf16);      // 32 MB
  constexpr size_t OFF_LMAX = OFF_L + SZ_L;
  constexpr size_t OFF_PO   = OFF_LMAX + 65536;
  constexpr size_t SZ_PO    = (size_t)NB * 16 * 64 * 64 * sizeof(float); // 8 MB
  constexpr size_t OFF_PSUM = OFF_PO + SZ_PO;

  bf16*  qls    = (bf16*)(w + OFF_QLS);
  bf16*  kl     = (bf16*)(w + OFF_KL);
  bf16*  k2invT = (bf16*)(w + OFF_K2);
  bf16*  k3T    = (bf16*)(w + OFF_K3);
  bf16*  vT     = (bf16*)(w + OFF_VT);
  bf16*  Lb     = (bf16*)(w + OFF_L);
  float* Lmax   = (float*)(w + OFF_LMAX);
  float* pO     = (float*)(w + OFF_PO);
  float* psum   = (float*)(w + OFF_PSUM);

  nys_pool  <<<NB * NL, 64, 0, stream>>>(q, k, qls, kl);
  nys_vtrans<<<dim3(SEQ / 128, NB), 256, 0, stream>>>(v, vT);
  nys_pinv  <<<NB, 256, 0, stream>>>(qls, kl, k2invT);
  nys_k3a   <<<dim3(SEQ / 512, NB), 256, 0, stream>>>(qls, k, Lb);
  nys_lmax  <<<NB * NL, 256, 0, stream>>>(Lb, Lmax);
  nys_k3b   <<<dim3(SEQ / 1024, NB), 256, 0, stream>>>(Lb, Lmax, vT, pO, psum);
  nys_k3c   <<<NB, 256, 0, stream>>>(pO, psum, k3T);
  nys_fuse  <<<dim3(SEQ / 128, NB), 256, 0, stream>>>(q, kl, k2invT, k3T, out);
}
